// MultiHeadAttention_71253507441351
// MI455X (gfx1250) — compile-verified, matrix operand order checked by reading, not run on hardware
//
#include <hip/hip_runtime.h>
#include <hip/hip_bf16.h>
#include <stdint.h>

// Problem constants (match reference)
#define BB  4
#define SS  2048
#define DD  1024
#define HH  16
#define DKK 64

typedef __bf16 bf16;
typedef __attribute__((ext_vector_type(16))) __bf16 v16bf;
typedef __attribute__((ext_vector_type(8)))  __bf16 v8bf;
typedef __attribute__((ext_vector_type(8)))  float  v8f;

// Native fptrunc (RNE) — lowers to hardware cvt.
static __device__ __forceinline__ bf16 f2bf(float f) {
  return static_cast<bf16>(f);
}

#define CAT16(lo, hi) \
  __builtin_shufflevector((lo), (hi), 0, 1, 2, 3, 4, 5, 6, 7, 8, 9, 10, 11, 12, 13, 14, 15)

// Load one 16x32 bf16 WMMA fragment from LDS (row-major, ldr elements/row).
// Lane l<16: row=l, K 0..7 & 16..23 ; lane l>=16: row=l-16, K 8..15 & 24..31.
static __device__ __forceinline__ v16bf ld_frag(const bf16* p, int ldr) {
  const int lane = threadIdx.x & 31;
  const int hf = lane >> 4, lr = lane & 15;
  const bf16* q = p + lr * ldr + hf * 8;
  v8bf lo = *(const v8bf*)(q);
  v8bf hi = *(const v8bf*)(q + 16);
  return CAT16(lo, hi);
}

static __device__ __forceinline__ v8f wmma_bf16(v16bf a, v16bf b, v8f c) {
  return __builtin_amdgcn_wmma_f32_16x16x32_bf16(false, a, false, b,
                                                 (short)0, c, false, false);
}

// ---------------------------------------------------------------------------
// NT GEMM: Y[m,n] = out_scale * sum_k A[m,k] * W[n,k]
// M = BB*SS = 8192, N = K = 1024. A either f32 or bf16; Y either bf16 or f32.
// Block = 128 threads = 4 waves (2x2); each wave owns a 64x64 tile.
// Software-pipelined: tile k+1 fetched to regs (converted to bf16 in flight)
// while WMMAs consume tile k from double-buffered LDS; tile k+2 prefetched.
// ---------------------------------------------------------------------------
template <bool A_F32, bool OUT_BF16>
__global__ __launch_bounds__(128) void gemm_nt_kernel(
    const void* __restrict__ Aptr, const float* __restrict__ W,
    void* __restrict__ Yptr, float out_scale) {
  constexpr int N = DD, K = DD;
  constexpr int BM = 128, BN = 128, BK = 32, LDT = 40;  // 80B row: conflict-free frags
  constexpr int NK = K / BK;
  __shared__ __align__(16) bf16 As[2][BM * LDT];
  __shared__ __align__(16) bf16 Bs[2][BN * LDT];

  const int tid = threadIdx.x;
  const int lane = tid & 31, wave = tid >> 5;
  const int hf = lane >> 4, lr = lane & 15;
  const int wm = (wave >> 1) * 64;   // wave's 64-row strip
  const int wn = (wave & 1) * 64;    // wave's 64-col strip
  const size_t m0 = (size_t)blockIdx.x * BM;
  const size_t n0 = (size_t)blockIdx.y * BN;

  const float* Af = (const float*)Aptr;
  const bf16*  Ab = (const bf16*)Aptr;

  v8f acc[4][4] = {};
  v8bf ar[4], br[4];                 // staged (bf16-converted) next tile

  // fetch tile at k0 into registers, converting to bf16 in flight
  auto fetch = [&](int k0) {
#pragma unroll
    for (int i = 0; i < 4; ++i) {
      int c = tid + i * 128;         // 512 chunks of 8 elements
      int row = c >> 2, col = (c & 3) * 8;
      if constexpr (A_F32) {
        const float* p = Af + (m0 + row) * (size_t)K + k0 + col;
        float4 lo = *(const float4*)(p);
        float4 hi = *(const float4*)(p + 4);
        v8bf t;
        t[0] = f2bf(lo.x); t[1] = f2bf(lo.y); t[2] = f2bf(lo.z); t[3] = f2bf(lo.w);
        t[4] = f2bf(hi.x); t[5] = f2bf(hi.y); t[6] = f2bf(hi.z); t[7] = f2bf(hi.w);
        ar[i] = t;
      } else {
        ar[i] = *(const v8bf*)(Ab + (m0 + row) * (size_t)K + k0 + col);
      }
      const float* q = W + (n0 + row) * (size_t)K + k0 + col;
      float4 lo = *(const float4*)(q);
      float4 hi = *(const float4*)(q + 4);
      v8bf t;
      t[0] = f2bf(lo.x); t[1] = f2bf(lo.y); t[2] = f2bf(lo.z); t[3] = f2bf(lo.w);
      t[4] = f2bf(hi.x); t[5] = f2bf(hi.y); t[6] = f2bf(hi.z); t[7] = f2bf(hi.w);
      br[i] = t;
    }
  };
  auto stage = [&](int buf) {
#pragma unroll
    for (int i = 0; i < 4; ++i) {
      int c = tid + i * 128;
      int row = c >> 2, col = (c & 3) * 8;
      *(v8bf*)&As[buf][row * LDT + col] = ar[i];
      *(v8bf*)&Bs[buf][row * LDT + col] = br[i];
    }
  };

  fetch(0);
  for (int kt = 0; kt < NK; ++kt) {
    const int buf = kt & 1;
    stage(buf);
    __syncthreads();
    if (kt + 1 < NK) fetch((kt + 1) * BK);          // overlaps with WMMAs below
    if (kt + 2 < NK) {                              // L2 prefetch of tile k+2
      const int pk = (kt + 2) * BK;
      if constexpr (A_F32) __builtin_prefetch(Af + (m0 + tid) * (size_t)K + pk, 0, 1);
      else                 __builtin_prefetch(Ab + (m0 + tid) * (size_t)K + pk, 0, 1);
      __builtin_prefetch(W + (n0 + tid) * (size_t)K + pk, 0, 1);
    }

    v16bf a[4], b[4];
#pragma unroll
    for (int mt = 0; mt < 4; ++mt) a[mt] = ld_frag(&As[buf][(wm + mt * 16) * LDT], LDT);
#pragma unroll
    for (int nt = 0; nt < 4; ++nt) b[nt] = ld_frag(&Bs[buf][(wn + nt * 16) * LDT], LDT);
#pragma unroll
    for (int mt = 0; mt < 4; ++mt)
#pragma unroll
      for (int nt = 0; nt < 4; ++nt)
        acc[mt][nt] = wmma_bf16(a[mt], b[nt], acc[mt][nt]);
  }

  // C layout: VGPR r -> M = r + 8*half ; N = lane&15
#pragma unroll
  for (int mt = 0; mt < 4; ++mt)
#pragma unroll
    for (int nt = 0; nt < 4; ++nt)
#pragma unroll
      for (int r = 0; r < 8; ++r) {
        size_t m = m0 + wm + mt * 16 + hf * 8 + r;
        size_t n = n0 + wn + nt * 16 + lr;
        float v = acc[mt][nt][r] * out_scale;
        if constexpr (OUT_BF16) ((bf16*)Yptr)[m * N + n] = f2bf(v);
        else                    ((float*)Yptr)[m * N + n] = v;
      }
}

// ---------------------------------------------------------------------------
// Flash attention: one block (4 waves) per 64 query rows per (b,h).
// Q pre-scaled by 1/sqrt(DK) at projection time. Causal + padding mask,
// online softmax, P@V via WMMA with V staged transposed in LDS.
// ---------------------------------------------------------------------------
__global__ __launch_bounds__(128) void attn_kernel(
    const bf16* __restrict__ Qb, const bf16* __restrict__ Kb,
    const bf16* __restrict__ Vb, const int* __restrict__ amask,
    bf16* __restrict__ Ctx) {
  constexpr int LDT = 72;                       // 144B row, 16B-aligned frags
  __shared__ __align__(16) bf16 Ks[64 * LDT];   // K tile   (sk x dk)
  __shared__ __align__(16) bf16 Vt[64 * LDT];   // V tile^T (dk x sk)
  __shared__ __align__(16) bf16 Ps[4][16 * LDT];// per-wave P strip (q x sk)

  const int tid = threadIdx.x;
  const int lane = tid & 31, wv = tid >> 5;
  const int hf = lane >> 4, lr = lane & 15;
  const int q0 = blockIdx.x * 64;
  const int bb = blockIdx.y / HH, hh = blockIdx.y % HH;
  const size_t rowbase = (size_t)bb * SS;
  const size_t hoff = (size_t)hh * DKK;

  // Resident Q strip: rows q0 + wv*16 .. +15, dk 0..63 -> two A fragments
  v16bf qa0, qa1;
  {
    const bf16* qp = Qb + (rowbase + q0 + wv * 16 + lr) * DD + hoff + hf * 8;
    v8bf x0 = *(const v8bf*)(qp);
    v8bf x1 = *(const v8bf*)(qp + 16);
    v8bf x2 = *(const v8bf*)(qp + 32);
    v8bf x3 = *(const v8bf*)(qp + 48);
    qa0 = CAT16(x0, x1);
    qa1 = CAT16(x2, x3);
  }

  v8f o[4] = {};
  float mrow[8], lrow[8];
#pragma unroll
  for (int r = 0; r < 8; ++r) { mrow[r] = -1e30f; lrow[r] = 0.f; }

  const int ntiles = blockIdx.x + 1;            // causal: k tiles <= q tile
  for (int kt = 0; kt < ntiles; ++kt) {
    const int k0 = kt * 64;
    __syncthreads();
    // stage K (straight) and V (transposed) tiles
#pragma unroll
    for (int i = 0; i < 4; ++i) {
      int c = tid + i * 128;
      int row = c >> 3, col = (c & 7) * 8;
      const size_t g = (rowbase + k0 + row) * DD + hoff + col;
      *(v8bf*)&Ks[row * LDT + col] = *(const v8bf*)(Kb + g);
      v8bf vvv = *(const v8bf*)(Vb + g);
#pragma unroll
      for (int j = 0; j < 8; ++j) Vt[(col + j) * LDT + row] = vvv[j];
    }
    // prefetch next K/V tile into GL2 while this one is consumed
    if (kt + 1 < ntiles) {
      const size_t gn = (rowbase + (size_t)(kt + 1) * 64 + (tid & 63)) * DD + hoff;
      __builtin_prefetch((tid < 64 ? Kb : Vb) + gn, 0, 1);
    }
    __syncthreads();

    // S strip = Q(16x64) . K^T  (4 col-tiles of 16)
    v8f sf[4];
#pragma unroll
    for (int nt = 0; nt < 4; ++nt) {
      v8f c = {};
      c = wmma_bf16(qa0, ld_frag(&Ks[nt * 16 * LDT], LDT), c);
      c = wmma_bf16(qa1, ld_frag(&Ks[nt * 16 * LDT + 32], LDT), c);
      sf[nt] = c;
    }
    // causal + padding mask
#pragma unroll
    for (int nt = 0; nt < 4; ++nt) {
      const int kg = k0 + nt * 16 + lr;
      const int ok = amask[(size_t)bb * SS + kg];
#pragma unroll
      for (int r = 0; r < 8; ++r) {
        const int qg = q0 + wv * 16 + hf * 8 + r;
        if (kg > qg || ok == 0) sf[nt][r] = -1e30f;
      }
    }
    // online softmax (row stats across 16 lanes of this half)
#pragma unroll
    for (int r = 0; r < 8; ++r) {
      float pm = fmaxf(fmaxf(sf[0][r], sf[1][r]), fmaxf(sf[2][r], sf[3][r]));
#pragma unroll
      for (int off = 1; off < 16; off <<= 1) pm = fmaxf(pm, __shfl_xor(pm, off, 32));
      const float mnew = fmaxf(mrow[r], pm);
      const float corr = __expf(mrow[r] - mnew);
      mrow[r] = mnew;
      float rs = 0.f;
#pragma unroll
      for (int nt = 0; nt < 4; ++nt) {
        const float p = __expf(sf[nt][r] - mnew);
        sf[nt][r] = p;
        rs += p;
      }
#pragma unroll
      for (int off = 1; off < 16; off <<= 1) rs += __shfl_xor(rs, off, 32);
      lrow[r] = lrow[r] * corr + rs;
#pragma unroll
      for (int nt = 0; nt < 4; ++nt) o[nt][r] *= corr;
    }
    // C-layout -> A-layout for P via wave-private LDS strip
#pragma unroll
    for (int nt = 0; nt < 4; ++nt)
#pragma unroll
      for (int r = 0; r < 8; ++r)
        Ps[wv][(hf * 8 + r) * LDT + nt * 16 + lr] = f2bf(sf[nt][r]);

    v16bf pa0 = ld_frag(&Ps[wv][0], LDT);
    v16bf pa1 = ld_frag(&Ps[wv][32], LDT);
#pragma unroll
    for (int nt = 0; nt < 4; ++nt) {
      o[nt] = wmma_bf16(pa0, ld_frag(&Vt[nt * 16 * LDT], LDT), o[nt]);
      o[nt] = wmma_bf16(pa1, ld_frag(&Vt[nt * 16 * LDT + 32], LDT), o[nt]);
    }
  }

  // normalize and write context (bf16, head-interleaved like Q/K/V)
#pragma unroll
  for (int r = 0; r < 8; ++r) {
    const float inv = 1.0f / lrow[r];
    const size_t s = rowbase + q0 + wv * 16 + hf * 8 + r;
#pragma unroll
    for (int nt = 0; nt < 4; ++nt)
      Ctx[s * DD + hoff + nt * 16 + lr] = f2bf(o[nt][r] * inv);
  }
}

// ---------------------------------------------------------------------------
extern "C" void kernel_launch(void* const* d_in, const int* in_sizes, int n_in,
                              void* d_out, int out_size, void* d_ws, size_t ws_size,
                              hipStream_t stream) {
  (void)in_sizes; (void)n_in; (void)out_size; (void)ws_size;
  const float* xq = (const float*)d_in[0];
  const float* xk = (const float*)d_in[1];
  const float* xv = (const float*)d_in[2];
  const int*   am = (const int*)d_in[3];
  const float* Wq = (const float*)d_in[4];
  const float* Wk = (const float*)d_in[5];
  const float* Wv = (const float*)d_in[6];
  const float* Wo = (const float*)d_in[7];

  const size_t BSD = (size_t)BB * SS * DD;   // 8,388,608 elems
  bf16* Qb  = (bf16*)d_ws;                   // ws usage: 4*BSD*2 = 64 MiB
  bf16* Kb  = Qb + BSD;
  bf16* Vb  = Kb + BSD;
  bf16* Ctx = Vb + BSD;

  dim3 ggrid((BB * SS) / 128, DD / 128);     // 64 x 8 blocks
  // Q projection with fused 1/sqrt(DK) softmax scale
  gemm_nt_kernel<true, true><<<ggrid, 128, 0, stream>>>(xq, Wq, Qb, 0.125f);
  gemm_nt_kernel<true, true><<<ggrid, 128, 0, stream>>>(xk, Wk, Kb, 1.0f);
  gemm_nt_kernel<true, true><<<ggrid, 128, 0, stream>>>(xv, Wv, Vb, 1.0f);

  attn_kernel<<<dim3(SS / 64, BB * HH), 128, 0, stream>>>(Qb, Kb, Vb, am, Ctx);

  gemm_nt_kernel<false, false><<<ggrid, 128, 0, stream>>>(Ctx, Wo, d_out, 1.0f);
}
